// GemmaAttention_73641509257614
// MI455X (gfx1250) — compile-verified, matrix operand order checked by reading, not run on hardware
//
#include <hip/hip_runtime.h>
#include <hip/hip_bf16.h>

// ---------------------------------------------------------------------------
// Gemma GQA attention for MI455X (gfx1250, wave32, WMMA + async-LDS staging).
//   NUM_HEADS=8, NUM_KV_HEADS=1, HEAD_DIM=256, B=4, S=2048, HID=2048.
// Pipeline (all on `stream`):
//   1) f32->f16 convert of hidden_states + 4 weight matrices
//   2) WMMA GEMM: Q/K/V projections (f16 out)
//   3) RoPE on Q (relayout to [b,h,s,d]) and K
//   4) WMMA GEMM: scores = Q K^T * 1/16 + mask  -> f32 into d_out attn_weights
//   5) row softmax in place on attn_weights
//   6) WMMA GEMM: P (f32, cvt on LDS fill) x V  -> f16 attn [b,s,h*d]
//   7) WMMA GEMM: attn x Wo -> f32 attn_output (d_out front)
// GEMM inner loop: LDS double-buffered, f16 tiles staged with
// global_load_async_to_lds_b128 (ASYNCcnt) so the next tile's HBM fetch
// overlaps the current tile's v_wmma work.
// ---------------------------------------------------------------------------

typedef __attribute__((ext_vector_type(16))) _Float16 v16h;
typedef __attribute__((ext_vector_type(8)))  _Float16 v8h;
typedef __attribute__((ext_vector_type(8)))  float    v8f;

#define NHEADS   8
#define SEQ      2048
#define HID      2048
#define HDIM     256
#define BATCH    4

// ---------------------------------------------------------------------------
__global__ __launch_bounds__(256)
void f32_to_f16_kernel(const float* __restrict__ src, _Float16* __restrict__ dst, long n) {
    long i = (long)blockIdx.x * 256 + threadIdx.x;
    if (i < n) dst[i] = (_Float16)src[i];
}

// ---------------------------------------------------------------------------
// RoPE: one block per (b,h,s); 128 threads, thread j handles dims j and j+128.
__global__ __launch_bounds__(128)
void rope_kernel(const _Float16* __restrict__ src, _Float16* __restrict__ dst,
                 int H, int S) {
    int idx = blockIdx.x;               // (b*H + h)*S + s
    int s  = idx % S;
    int bh = idx / S;
    int h  = bh % H;
    int b  = bh / H;
    int j  = threadIdx.x;               // 0..127
    float inv_freq = __expf((float)j * (-9.210340371976184f / 128.0f)); // 10000^(-j/128)
    float fr = (float)s * inv_freq;
    float c  = cosf(fr);
    float sn = sinf(fr);
    const _Float16* row = src + (long)(b * S + s) * (H * HDIM) + h * HDIM;
    _Float16* orow = dst + (long)idx * HDIM;
    float x1 = (float)row[j];
    float x2 = (float)row[j + 128];
    orow[j]       = (_Float16)(x1 * c - x2 * sn);
    orow[j + 128] = (_Float16)(x2 * c + x1 * sn);
}

// ---------------------------------------------------------------------------
// Row softmax, one 256-thread block per row of 2048 f32, in place.
__global__ __launch_bounds__(256)
void softmax_rows_kernel(float* __restrict__ data) {
    long row = blockIdx.x;
    float* p = data + row * (long)SEQ;
    int tid = threadIdx.x;
    float v[8];
    float mx = -3.4e38f;
#pragma unroll
    for (int i = 0; i < 8; ++i) { v[i] = p[tid + i * 256]; mx = fmaxf(mx, v[i]); }
    __shared__ float red[256];
    red[tid] = mx; __syncthreads();
    for (int s = 128; s > 0; s >>= 1) {
        if (tid < s) red[tid] = fmaxf(red[tid], red[tid + s]);
        __syncthreads();
    }
    mx = red[0]; __syncthreads();
    float sum = 0.f;
#pragma unroll
    for (int i = 0; i < 8; ++i) { v[i] = __expf(v[i] - mx); sum += v[i]; }
    red[tid] = sum; __syncthreads();
    for (int s = 128; s > 0; s >>= 1) {
        if (tid < s) red[tid] += red[tid + s];
        __syncthreads();
    }
    float inv = 1.0f / red[0];
#pragma unroll
    for (int i = 0; i < 8; ++i) p[tid + i * 256] = v[i] * inv;
}

// ---------------------------------------------------------------------------
// Async 16-byte global -> LDS copy (gfx1250, tracked by ASYNCcnt).
__device__ __forceinline__ void async_copy16(const void* gsrc, void* ldst) {
    unsigned lds = (unsigned)(size_t)ldst;              // low 32 bits = LDS address
    unsigned long long ga = (unsigned long long)(size_t)gsrc;
    asm volatile("global_load_async_to_lds_b128 %0, %1, off"
                 :: "v"(lds), "v"(ga) : "memory");
}
__device__ __forceinline__ void wait_async0() {
    asm volatile("s_wait_asynccnt 0x0" ::: "memory");
}

// ---------------------------------------------------------------------------
// Tiled WMMA GEMM: C[M,N] = epilogue(A[M,K] * B[K,N]).
//   Block tile 128x64, BK=32, 256 threads = 8 waves, each wave a 32x32 tile
//   (2x2 v_wmma_f32_16x16x32_f16).  LDS double-buffered; f16 tiles are staged
//   with global_load_async_to_lds_b128 so tile t+1 streams in during tile t's
//   WMMA work.  A source may be f16 (async) or f32 (converted on fill, P@V).
//   B row-major [K,N] (ds scatter into K-major LDS) or TRANSB source [N,K]
//   (async straight copy).  Bs is K-major ([n][k]) so both fragments load as
//   two 16-B LDS vectors per lane, matching the ISA 16-bit 16x32 layout:
//     lane<16 : K in {0..7, 16..23}; lane>=16 : K in {8..15, 24..31}.
//   EPI==1: C = acc*scale + mask[r,c] (scores); else plain store.
// ---------------------------------------------------------------------------
template<typename AT, bool TRANSB, typename OT, int EPI>
__global__ __launch_bounds__(256)
void gemm_wmma_kernel(const AT* __restrict__ Ab, const _Float16* __restrict__ Bb,
                      OT* __restrict__ Cb, const float* __restrict__ Maskb,
                      int K, int lda, int ldb, int ldc,
                      long aStride, long bStride, int bDiv,
                      long cStrideO, long cStrideI, int cDiv,
                      long mStride, int mDiv, int ldm, float scale) {
    __shared__ __align__(16) _Float16 As[2][128][40];
    __shared__ __align__(16) _Float16 Bs[2][64][40];

    const int z = blockIdx.z;
    const AT* A = Ab + (long)z * aStride;
    const _Float16* B = Bb + (long)(z / bDiv) * bStride;
    OT* C = Cb + (long)(z / cDiv) * cStrideO + (long)(z % cDiv) * cStrideI;
    const float* Mask = nullptr;
    if constexpr (EPI == 1) Mask = Maskb + (long)(z / mDiv) * mStride;

    const int m0 = blockIdx.y * 128;
    const int n0 = blockIdx.x * 64;
    const int tid = threadIdx.x;
    const int lane = tid & 31;
    const int wave = tid >> 5;
    const int wm = (wave & 3) * 32;   // wave M offset in block tile
    const int wn = (wave >> 2) * 32;  // wave N offset in block tile

    const int nT = K / 32;            // all K's are multiples of 32

    // stage tile t of A and B into LDS buffer bufi
    auto issue = [&](int t, int bufi) {
        const int kb = t * 32;
        {   // ---- A tile (128x32): 16 elems per thread ----
            const int row = tid >> 1;
            const int cg  = (tid & 1) * 16;
            if constexpr (sizeof(AT) == 2) {
                const _Float16* src = (const _Float16*)A + (long)(m0 + row) * lda + kb + cg;
                async_copy16(src,     &As[bufi][row][cg]);
                async_copy16(src + 8, &As[bufi][row][cg + 8]);
            } else {   // f32 source: convert while filling (P@V path)
                const float* src = (const float*)A + (long)(m0 + row) * lda + kb + cg;
#pragma unroll
                for (int i = 0; i < 16; i += 4) {
                    float4 f = *(const float4*)(src + i);
                    As[bufi][row][cg + i + 0] = (_Float16)f.x;
                    As[bufi][row][cg + i + 1] = (_Float16)f.y;
                    As[bufi][row][cg + i + 2] = (_Float16)f.z;
                    As[bufi][row][cg + i + 3] = (_Float16)f.w;
                }
            }
        }
        if constexpr (TRANSB) {        // B source is [N,K] row-major: direct async copy
            const int ncol = tid >> 2;     // 0..63
            const int kk   = (tid & 3) * 8;
            const _Float16* src = B + (long)(n0 + ncol) * ldb + kb + kk;
            async_copy16(src, &Bs[bufi][ncol][kk]);
        } else {                       // B source is [K,N] row-major: transpose into Bs[n][k]
            const int krow = tid >> 3;     // 0..31
            const int col  = (tid & 7) * 8;
            const _Float16* src = B + (long)(kb + krow) * ldb + n0 + col;
            v8h vb = *(const v8h*)src;
#pragma unroll
            for (int j = 0; j < 8; ++j) Bs[bufi][col + j][krow] = vb[j];
        }
    };

    v8f acc[2][2] = {};

    issue(0, 0);
    for (int t = 0; t < nT; ++t) {
        wait_async0();        // buffer (t&1) fully landed for this wave
        __syncthreads();      // ... and for every wave; prior readers done
        if (t + 1 < nT) issue(t + 1, (t + 1) & 1);

        const int bufc = t & 1;
        const int halfk = lane >> 4;
        const int ml = lane & 15;
        v16h af[2], bf[2];
#pragma unroll
        for (int tt = 0; tt < 2; ++tt) {
            const _Float16* pa = &As[bufc][wm + tt * 16 + ml][8 * halfk];
            v8h lo = *(const v8h*)pa;
            v8h hi = *(const v8h*)(pa + 16);
#pragma unroll
            for (int e = 0; e < 8; ++e) { af[tt][e] = lo[e]; af[tt][8 + e] = hi[e]; }
            const _Float16* pb = &Bs[bufc][wn + tt * 16 + ml][8 * halfk];
            v8h blo = *(const v8h*)pb;
            v8h bhi = *(const v8h*)(pb + 16);
#pragma unroll
            for (int e = 0; e < 8; ++e) { bf[tt][e] = blo[e]; bf[tt][8 + e] = bhi[e]; }
        }
#pragma unroll
        for (int i = 0; i < 2; ++i)
#pragma unroll
            for (int j = 0; j < 2; ++j)
                acc[i][j] = __builtin_amdgcn_wmma_f32_16x16x32_f16(
                    false, af[i], false, bf[j], (short)0, acc[i][j], false, false);
    }

    // ---- epilogue: C/D layout — VGPR v -> M = v + 8*(lane>=16), N = lane&15
    const int halfc = lane >> 4;
    const int nl = lane & 15;
#pragma unroll
    for (int ti = 0; ti < 2; ++ti)
#pragma unroll
        for (int tj = 0; tj < 2; ++tj)
#pragma unroll
            for (int v = 0; v < 8; ++v) {
                const int r = m0 + wm + ti * 16 + v + 8 * halfc;
                const int c = n0 + wn + tj * 16 + nl;
                float val = acc[ti][tj][v];
                if constexpr (EPI == 1)
                    val = val * scale + Mask[(long)r * ldm + c];
                C[(long)r * ldc + c] = (OT)val;
            }
}

// ---------------------------------------------------------------------------
extern "C" void kernel_launch(void* const* d_in, const int* in_sizes, int n_in,
                              void* d_out, int out_size, void* d_ws, size_t ws_size,
                              hipStream_t stream) {
    (void)in_sizes; (void)n_in; (void)out_size; (void)ws_size;

    const float* hs   = (const float*)d_in[0];  // [4,2048,2048]
    const float* mask = (const float*)d_in[1];  // [4,1,2048,2048]
    const float* wq   = (const float*)d_in[2];  // [2048,2048]
    const float* wk   = (const float*)d_in[3];  // [2048,256]
    const float* wv   = (const float*)d_in[4];  // [2048,256]
    const float* wo   = (const float*)d_in[5];  // [2048,2048]

    float* out0  = (float*)d_out;                       // attn_output [4,2048,2048]
    float* attnw = out0 + (long)BATCH * SEQ * HID;      // attn_weights [4,8,2048,2048]

    // workspace layout (bytes)
    char* w = (char*)d_ws;
    _Float16* hs16    = (_Float16*)(w + 0);             // 8192x2048
    _Float16* wq16    = (_Float16*)(w + 33554432);      // 2048x2048
    _Float16* wk16    = (_Float16*)(w + 41943040);      // 2048x256
    _Float16* wv16    = (_Float16*)(w + 42991616);      // 2048x256
    _Float16* wo16    = (_Float16*)(w + 44040192);      // 2048x2048
    _Float16* qproj16 = (_Float16*)(w + 52428800);      // 8192x2048 (pre-RoPE Q)
    _Float16* kproj16 = (_Float16*)(w + 85983232);      // 8192x256  (pre-RoPE K)
    _Float16* q16     = (_Float16*)(w + 90177536);      // [b,h,s,d] post-RoPE
    _Float16* k16     = (_Float16*)(w + 123731968);     // [b,s,d]   post-RoPE
    _Float16* v16     = (_Float16*)(w + 127926272);     // [b,s,d]
    _Float16* attn16  = qproj16;                        // reuse: [b,s,h*d]

    // 1) f32 -> f16 conversions
    f32_to_f16_kernel<<<65536, 256, 0, stream>>>(hs, hs16, (long)8192 * HID);
    f32_to_f16_kernel<<<16384, 256, 0, stream>>>(wq, wq16, (long)HID * HID);
    f32_to_f16_kernel<<<2048,  256, 0, stream>>>(wk, wk16, (long)HID * HDIM);
    f32_to_f16_kernel<<<2048,  256, 0, stream>>>(wv, wv16, (long)HID * HDIM);
    f32_to_f16_kernel<<<16384, 256, 0, stream>>>(wo, wo16, (long)HID * HID);

    // 2) projections: [8192,2048] x W
    gemm_wmma_kernel<_Float16, false, _Float16, 0><<<dim3(HID / 64, 8192 / 128, 1), 256, 0, stream>>>(
        hs16, wq16, qproj16, nullptr, HID, HID, HID, HID,
        0, 0, 1, 0, 0, 1, 0, 1, 0, 1.0f);
    gemm_wmma_kernel<_Float16, false, _Float16, 0><<<dim3(HDIM / 64, 8192 / 128, 1), 256, 0, stream>>>(
        hs16, wk16, kproj16, nullptr, HID, HID, HDIM, HDIM,
        0, 0, 1, 0, 0, 1, 0, 1, 0, 1.0f);
    gemm_wmma_kernel<_Float16, false, _Float16, 0><<<dim3(HDIM / 64, 8192 / 128, 1), 256, 0, stream>>>(
        hs16, wv16, v16, nullptr, HID, HID, HDIM, HDIM,
        0, 0, 1, 0, 0, 1, 0, 1, 0, 1.0f);

    // 3) RoPE
    rope_kernel<<<BATCH * NHEADS * SEQ, 128, 0, stream>>>(qproj16, q16, NHEADS, SEQ);
    rope_kernel<<<BATCH * SEQ,          128, 0, stream>>>(kproj16, k16, 1, SEQ);

    // 4) scores = Q K^T / 16 + mask   (z = b*8+h, 32 batched GEMMs)
    gemm_wmma_kernel<_Float16, true, float, 1><<<dim3(SEQ / 64, SEQ / 128, 32), 256, 0, stream>>>(
        q16, k16, attnw, mask, HDIM, HDIM, HDIM, SEQ,
        (long)SEQ * HDIM,           // A: per (b,h) Q slice
        (long)SEQ * HDIM, NHEADS,   // B: per b K slice
        (long)SEQ * SEQ, 0, 1,      // C: per z slice of attn_weights
        (long)SEQ * SEQ, NHEADS, SEQ, 0.0625f);

    // 5) softmax in place on attn_weights
    softmax_rows_kernel<<<BATCH * NHEADS * SEQ, 256, 0, stream>>>(attnw);

    // 6) attn = P x V  (P f32 converted on LDS fill) -> f16 [b,s,h*d]
    gemm_wmma_kernel<float, false, _Float16, 0><<<dim3(HDIM / 64, SEQ / 128, 32), 256, 0, stream>>>(
        attnw, v16, attn16, nullptr, SEQ, SEQ, HDIM, HID,
        (long)SEQ * SEQ,            // A: per z P slice
        (long)SEQ * HDIM, NHEADS,   // B: per b V slice
        (long)SEQ * HID, HDIM, NHEADS,  // C: base b*S*2048 + h*256
        0, 1, 0, 1.0f);

    // 7) out = attn x Wo -> f32
    gemm_wmma_kernel<_Float16, false, float, 0><<<dim3(HID / 64, 8192 / 128, 1), 256, 0, stream>>>(
        attn16, wo16, out0, nullptr, HID, HID, HID, HID,
        0, 0, 1, 0, 0, 1, 0, 1, 0, 1.0f);
}
